// FusedFP8Linear_43001212567659
// MI455X (gfx1250) — 8-bit matrix operand order repaired and checked by reading, compiled, not run on hardware
//
#include <hip/hip_runtime.h>
#include <hip/hip_bf16.h>
#include <stdint.h>

typedef __attribute__((ext_vector_type(16))) int   v16i;
typedef __attribute__((ext_vector_type(8)))  float v8f;

#define TILE      128
#define KCHUNK    128
#define LDSTRIDE  128

#if __has_builtin(__builtin_amdgcn_cvt_pk_fp8_f32)
#define HAVE_HW_FP8 1
#else
#define HAVE_HW_FP8 0
#endif

__device__ __forceinline__ float bf16_to_f32(unsigned short u) {
  return __uint_as_float(((unsigned)u) << 16);
}
__device__ __forceinline__ unsigned short f32_to_bf16(float f) {
  unsigned u = __float_as_uint(f);
  return (unsigned short)((u + 0x7FFFu + ((u >> 16) & 1u)) >> 16);
}
__device__ __forceinline__ unsigned f32_to_e4m3_sw(float f) {
  unsigned u = __float_as_uint(f);
  unsigned sign = (u >> 24) & 0x80u;
  u &= 0x7FFFFFFFu;
  float a = __uint_as_float(u);
  if (a < 0.015625f) {
    int q = (int)rintf(a * 512.0f);
    return sign | (unsigned)q;
  }
  u += 0x7FFFFu + ((u >> 20) & 1u);
  int e8 = (int)(u >> 23) - 120;
  if (e8 > 15) return sign | 0x7Eu;
  return sign | ((unsigned)e8 << 3) | ((u >> 20) & 0x7u);
}
__device__ __forceinline__ float clamp448(float f) {
  return fminf(448.0f, fmaxf(-448.0f, f));
}

__global__ void quant_e4m3_kernel(const unsigned short* __restrict__ in,
                                  unsigned char* __restrict__ out,
                                  const float* __restrict__ scale,
                                  long long n16) {
  long long i = (long long)blockIdx.x * blockDim.x + threadIdx.x;
  if (i >= n16) return;
  const float s = scale[0];
  const uint4* p = (const uint4*)in;
  uint4 d0 = p[2 * i];
  uint4 d1 = p[2 * i + 1];
  unsigned w[8] = {d0.x, d0.y, d0.z, d0.w, d1.x, d1.y, d1.z, d1.w};
  unsigned res[4];
#pragma unroll
  for (int k = 0; k < 4; ++k) {
    float f0 = clamp448(bf16_to_f32((unsigned short)(w[2 * k] & 0xFFFFu)) * s);
    float f1 = clamp448(bf16_to_f32((unsigned short)(w[2 * k] >> 16)) * s);
    float f2 = clamp448(bf16_to_f32((unsigned short)(w[2 * k + 1] & 0xFFFFu)) * s);
    float f3 = clamp448(bf16_to_f32((unsigned short)(w[2 * k + 1] >> 16)) * s);
#if HAVE_HW_FP8
    int t = __builtin_amdgcn_cvt_pk_fp8_f32(f0, f1, 0, false);
    res[k] = (unsigned)__builtin_amdgcn_cvt_pk_fp8_f32(f2, f3, t, true);
#else
    res[k] = f32_to_e4m3_sw(f0) | (f32_to_e4m3_sw(f1) << 8) |
             (f32_to_e4m3_sw(f2) << 16) | (f32_to_e4m3_sw(f3) << 24);
#endif
  }
  uint4 o; o.x = res[0]; o.y = res[1]; o.z = res[2]; o.w = res[3];
  ((uint4*)out)[i] = o;
}

__device__ __forceinline__ void async_b128(unsigned lds_off, const void* gaddr) {
  asm volatile("global_load_async_to_lds_b128 %0, %1, off"
               :: "v"(lds_off), "v"((unsigned long long)(uintptr_t)gaddr)
               : "memory");
}
__device__ __forceinline__ void wait_async0() {
#if __has_builtin(__builtin_amdgcn_s_wait_asynccnt)
  __builtin_amdgcn_s_wait_asynccnt(0);
#else
  asm volatile("s_wait_asynccnt 0x0" ::: "memory");
#endif
}

__launch_bounds__(256)
__global__ void fp8_gemm_bias_kernel(const unsigned char* __restrict__ Aq,
                                     const unsigned char* __restrict__ Bq,
                                     const unsigned short* __restrict__ bias,
                                     const float* __restrict__ xs,
                                     const float* __restrict__ ws,
                                     unsigned short* __restrict__ out,
                                     int M, int N, int K) {
  __shared__ __align__(16) unsigned char lsA[2][TILE * LDSTRIDE];
  __shared__ __align__(16) unsigned char lsB[2][TILE * LDSTRIDE];

  const int tid    = threadIdx.x;
  const int lane   = tid & 31;
  const int lane16 = lane & 15;
  const int hi     = lane >> 4;
  const int wid    = tid >> 5;

  const int m0 = (wid & 3) * 32;
  const int n0 = (wid >> 2) * 64;

  const long long blockM = (long long)blockIdx.y * TILE;
  const long long blockN = (long long)blockIdx.x * TILE;
  const unsigned char* gA = Aq + blockM * (long long)K;
  const unsigned char* gB = Bq + blockN * (long long)K;

  auto copy_tile = [&](int buf, int kt) {
    const long long kbase = (long long)kt * KCHUNK;
#pragma unroll
    for (int j = 0; j < 4; ++j) {
      int c   = tid + 256 * j;
      int row = c >> 3;
      int col = (c & 7) << 4;
      unsigned loff = (unsigned)(c << 4);
      async_b128((unsigned)(uintptr_t)&lsA[buf][loff],
                 gA + (long long)row * K + kbase + col);
      async_b128((unsigned)(uintptr_t)&lsB[buf][loff],
                 gB + (long long)row * K + kbase + col);
    }
  };

  v8f acc[2][4];
#pragma unroll
  for (int mt = 0; mt < 2; ++mt)
#pragma unroll
    for (int nt = 0; nt < 4; ++nt)
#pragma unroll
      for (int v = 0; v < 8; ++v) acc[mt][nt][v] = 0.0f;

  copy_tile(0, 0);
  wait_async0();
  __syncthreads();

  const int KT = K / KCHUNK;
  for (int kt = 0; kt < KT; ++kt) {
    const int buf = kt & 1;
    if (kt + 1 < KT) copy_tile(buf ^ 1, kt + 1);

    v16i afr[2];
#pragma unroll
    for (int mt = 0; mt < 2; ++mt) {
      const unsigned char* base =
          &lsA[buf][(m0 + mt * 16 + lane16) * LDSTRIDE + hi * 16];
#pragma unroll
      for (int p = 0; p < 8; ++p) {
        int off = ((p >> 1) << 5) + ((p & 1) << 3);
        int2 d = *(const int2*)(base + off);
        afr[mt][2 * p]     = d.x;
        afr[mt][2 * p + 1] = d.y;
      }
    }

#pragma unroll
    for (int nt = 0; nt < 4; ++nt) {
      v16i bfr;
      const unsigned char* base =
          &lsB[buf][(n0 + nt * 16 + lane16) * LDSTRIDE + hi * 16];
#pragma unroll
      for (int c4 = 0; c4 < 4; ++c4) {
        int4 d = *(const int4*)(base + c4 * 32);
        bfr[4 * c4]     = d.x;
        bfr[4 * c4 + 1] = d.y;
        bfr[4 * c4 + 2] = d.z;
        bfr[4 * c4 + 3] = d.w;
      }
      acc[0][nt] = __builtin_amdgcn_wmma_f32_16x16x128_fp8_fp8(
          afr[0], bfr, (short)0, acc[0][nt], false, false);
      acc[1][nt] = __builtin_amdgcn_wmma_f32_16x16x128_fp8_fp8(
          afr[1], bfr, (short)0, acc[1][nt], false, false);
    }

    wait_async0();
    __syncthreads();
  }

  unsigned short* stage = (unsigned short*)&lsA[0][0];
  const float inv = 1.0f / (xs[0] * ws[0]);
#pragma unroll
  for (int nt = 0; nt < 4; ++nt) {
    const int col = n0 + nt * 16 + lane16;
    const float bc = bf16_to_f32(bias[blockN + col]);
#pragma unroll
    for (int mt = 0; mt < 2; ++mt) {
      const int rbase = m0 + mt * 16 + hi * 8;
#pragma unroll
      for (int v = 0; v < 8; ++v) {
        stage[(rbase + v) * TILE + col] =
            f32_to_bf16(acc[mt][nt][v] * inv + bc);
      }
    }
  }
  __syncthreads();

#pragma unroll
  for (int j = 0; j < 8; ++j) {
    int c   = tid + 256 * j;
    int row = c >> 4;
    int col = (c & 15) << 3;
    uint4 d = *(const uint4*)&stage[row * TILE + col];
    *(uint4*)&out[(blockM + row) * (long long)N + blockN + col] = d;
  }
}

extern "C" void kernel_launch(void* const* d_in, const int* in_sizes, int n_in,
                              void* d_out, int out_size, void* d_ws, size_t ws_size,
                              hipStream_t stream) {
  (void)n_in; (void)out_size; (void)ws_size;
  const unsigned short* x    = (const unsigned short*)d_in[0];
  const unsigned short* wgt  = (const unsigned short*)d_in[1];
  const unsigned short* bias = (const unsigned short*)d_in[2];
  const float* xs = (const float*)d_in[3];
  const float* ws = (const float*)d_in[4];

  const long long N = in_sizes[2];
  const long long K = (long long)in_sizes[1] / N;
  const long long M = (long long)in_sizes[0] / K;

  unsigned char* xq = (unsigned char*)d_ws;
  unsigned char* wq = xq + (size_t)(M * K);

  const long long n16x = (M * K) / 16;
  const long long n16w = (N * K) / 16;
  quant_e4m3_kernel<<<(unsigned)((n16x + 255) / 256), 256, 0, stream>>>(x, xq, xs, n16x);
  quant_e4m3_kernel<<<(unsigned)((n16w + 255) / 256), 256, 0, stream>>>(wgt, wq, ws, n16w);

  dim3 grid((unsigned)(N / TILE), (unsigned)(M / TILE));
  fp8_gemm_bias_kernel<<<grid, 256, 0, stream>>>(xq, wq, bias, xs, ws,
                                                 (unsigned short*)d_out,
                                                 (int)M, (int)N, (int)K);
}
